// EBT_89266600280107
// MI455X (gfx1250) — compile-verified, matrix operand order checked by reading, not run on hardware
//
#include <hip/hip_runtime.h>
#include <stdint.h>

#define D_   1152
#define T_   256
#define NB   32
#define HH   16
#define HD_  72
#define HDP  96
#define DF_  4608
#define LL   12
#define ROWS (NB*T_)   // 8192

typedef __bf16 bf16x16 __attribute__((ext_vector_type(16)));
typedef float  f32x8   __attribute__((ext_vector_type(8)));
typedef __attribute__((address_space(3))) uint8_t lds_u8;

union Frag { uint32_t u[8]; uint4 q[2]; bf16x16 v; };

__device__ __forceinline__ uint16_t f2bf(float f) {
    uint32_t u = __float_as_uint(f);
    u += 0x7FFFu + ((u >> 16) & 1u);          // round-to-nearest-even
    return (uint16_t)(u >> 16);
}
__device__ __forceinline__ float bf2f(uint16_t h) {
    return __uint_as_float(((uint32_t)h) << 16);
}

template<typename T> __device__ __forceinline__ void storeOut(T* p, float v);
template<> __device__ __forceinline__ void storeOut<float>(float* p, float v) { *p = v; }
template<> __device__ __forceinline__ void storeOut<uint16_t>(uint16_t* p, float v) { *p = f2bf(v); }

__device__ __forceinline__ float wave_sum(float v) {
#pragma unroll
    for (int o = 16; o > 0; o >>= 1) v += __shfl_xor(v, o, 32);
    return v;
}
__device__ __forceinline__ float wave_max(float v) {
#pragma unroll
    for (int o = 16; o > 0; o >>= 1) v = fmaxf(v, __shfl_xor(v, o, 32));
    return v;
}

// CDNA5 async global->LDS copy, 16B per lane, tracked by ASYNCcnt.
__device__ __forceinline__ void async_load_b128(uint32_t lds_addr, const void* gaddr) {
    asm volatile("global_load_async_to_lds_b128 %0, %1, off"
                 :: "v"(lds_addr), "v"((uint64_t)(uintptr_t)gaddr)
                 : "memory");
}

// ---------------------------------------------------------------------------
// Generic bf16 WMMA GEMM: C[M,N] = A[M,K] @ B[N,K]^T + bias[N]
// Block = 128 threads = 4 waves; block tile = 128 rows x 64 cols.
// Each wave owns two 16-row tiles (8 f32 accumulators). The 64x32 B tile
// (shared by all waves / both row tiles) is staged in LDS via async copies,
// double buffered. M >= 32 and multiple of 16; N multiple of 16; K multiple
// of 32. Batched via blockIdx.z.
// ---------------------------------------------------------------------------
template<typename OutT>
__global__ __launch_bounds__(128)
void wmma_gemm(const uint16_t* __restrict__ A, int lda, long long sA,
               const uint16_t* __restrict__ B, int ldb, long long sB,
               OutT* __restrict__ C, int ldc, long long sC,
               const float* __restrict__ bias,
               int M, int N, int K)
{
    __shared__ uint8_t ldsB[2][64][64];       // [buf][n-row][32 K * bf16]

    const int lane = threadIdx.x & 31;
    const int wave = threadIdx.x >> 5;
    const int t    = threadIdx.x;
    const long long b = blockIdx.z;
    A += b * sA;  B += b * sB;  C += b * sC;

    int row0 = blockIdx.y * 128 + wave * 32;
    if (row0 > M - 32) row0 = M - 32;         // clamp (duplicate work, same result)
    const int col0 = blockIdx.x * 64;
    const int r  = lane & 15;
    const int hf = lane >> 4;

    const uint32_t lbase = (uint32_t)(uintptr_t)(lds_u8*)&ldsB[0][0][0];
    const uint8_t* gB = (const uint8_t*)B;

    // stage one 64x32 B tile (4KB) into buffer p: 256 x 16B chunks, 2/thread
    auto stage = [&](int k0, int p) {
#pragma unroll
        for (int j = 0; j < 2; j++) {
            int ci  = t + j * 128;            // 0..255
            int row = ci >> 2, sub = ci & 3;
            int nrow = col0 + row; if (nrow > N - 1) nrow = N - 1;
            const uint8_t* g = gB + ((long long)nrow * ldb + k0) * 2 + sub * 16;
            async_load_b128(lbase + (uint32_t)(p * 4096 + row * 64 + sub * 16), g);
        }
    };

    f32x8 acc0[4], acc1[4];
#pragma unroll
    for (int i = 0; i < 4; i++)
#pragma unroll
        for (int j = 0; j < 8; j++) { acc0[i][j] = 0.f; acc1[i][j] = 0.f; }

    const uint16_t* ap0 = A + (long long)(row0 + r) * lda;
    const uint16_t* ap1 = A + (long long)(row0 + 16 + r) * lda;

    auto compute = [&](int k0, int p) {
        Frag a0, a1;
        const uint4* pa0 = (const uint4*)(ap0 + k0);
        const uint4* pa1 = (const uint4*)(ap1 + k0);
        a0.q[0] = pa0[hf];  a0.q[1] = pa0[2 + hf];
        a1.q[0] = pa1[hf];  a1.q[1] = pa1[2 + hf];
        Frag bq[4];
#pragma unroll
        for (int nt = 0; nt < 4; nt++) {
            const uint4* pb = (const uint4*)&ldsB[p][nt * 16 + r][0];
            bq[nt].q[0] = pb[hf];
            bq[nt].q[1] = pb[2 + hf];
        }
#pragma unroll
        for (int nt = 0; nt < 4; nt++) {
            acc0[nt] = __builtin_amdgcn_wmma_f32_16x16x32_bf16(
                false, a0.v, false, bq[nt].v, (short)0, acc0[nt], false, false);
            acc1[nt] = __builtin_amdgcn_wmma_f32_16x16x32_bf16(
                false, a1.v, false, bq[nt].v, (short)0, acc1[nt], false, false);
        }
    };

    stage(0, 0);
    int p = 0;
    int k0 = 0;
    for (; k0 < K - 32; k0 += 32) {
        stage(k0 + 32, p ^ 1);
        asm volatile("s_wait_asynccnt 0x2" ::: "memory");  // oldest batch landed
        __syncthreads();                       // B tile p visible to all waves
        compute(k0, p);
        __syncthreads();                       // all waves done with buffer p
        p ^= 1;
    }
    asm volatile("s_wait_asynccnt 0x0" ::: "memory");
    __syncthreads();
    compute(k0, p);

#pragma unroll
    for (int nt = 0; nt < 4; nt++) {
        int c0 = col0 + nt * 16;
        if (c0 >= N) break;                    // store guard only (uniform)
        int col = c0 + r;
        float bv = bias ? bias[col] : 0.f;
#pragma unroll
        for (int rr = 0; rr < 8; rr++) {
            int m0 = row0 + rr + hf * 8;
            storeOut<OutT>(&C[(long long)m0 * ldc + col], acc0[nt][rr] + bv);
            storeOut<OutT>(&C[(long long)(m0 + 16) * ldc + col], acc1[nt][rr] + bv);
        }
    }
}

// ---------------------------------------------------------------------------
// Elementwise / helper kernels
// ---------------------------------------------------------------------------
__global__ void add_pos_kernel(const float* __restrict__ x, const float* __restrict__ pos,
                               float* __restrict__ xf, long long n) {
    long long i = (long long)blockIdx.x * 256 + threadIdx.x;
    if (i < n) xf[i] = x[i] + pos[i % ((long long)T_ * D_)];
}

__global__ void silu_bf_kernel(const float* __restrict__ y, uint16_t* __restrict__ out, long long n) {
    long long i = (long long)blockIdx.x * 256 + threadIdx.x;
    if (i < n) { float v = y[i]; out[i] = f2bf(v / (1.f + __expf(-v))); }
}

__global__ void cvt_kernel(const float* __restrict__ in, uint16_t* __restrict__ out, long long n) {
    long long i = (long long)blockIdx.x * 256 + threadIdx.x;
    if (i < n) out[i] = f2bf(in[i]);
}

__global__ void fill_zero_bf(uint16_t* __restrict__ p, long long n) {
    long long i = (long long)blockIdx.x * 256 + threadIdx.x;
    if (i < n) p[i] = 0;
}

template<typename OutT>
__global__ __launch_bounds__(256)
void ln_modulate(const float* __restrict__ x, const float* __restrict__ mod,
                 int modStride, int shOff, int scOff, OutT* __restrict__ out)
{
    int row = blockIdx.x;
    int n   = row >> 8;                       // row / T_
    const float* xr = x + (long long)row * D_;
    float s = 0.f, s2 = 0.f;
    for (int i = threadIdx.x; i < D_; i += 256) { float v = xr[i]; s += v; s2 += v * v; }
    s = wave_sum(s); s2 = wave_sum(s2);
    __shared__ float sb[16];
    if ((threadIdx.x & 31) == 0) { sb[threadIdx.x >> 5] = s; sb[8 + (threadIdx.x >> 5)] = s2; }
    __syncthreads();
    float S = 0.f, S2 = 0.f;
#pragma unroll
    for (int i = 0; i < 8; i++) { S += sb[i]; S2 += sb[8 + i]; }
    float mean = S * (1.f / D_);
    float var  = S2 * (1.f / D_) - mean * mean;
    float inv  = rsqrtf(var + 1e-6f);
    for (int i = threadIdx.x; i < D_; i += 256) {
        float v = (xr[i] - mean) * inv;
        if (mod) {
            float sc = mod[(long long)n * modStride + scOff + i];
            float sh = mod[(long long)n * modStride + shOff + i];
            v = v * (1.f + sc) + sh;
        }
        storeOut<OutT>(&out[(long long)row * D_ + i], v);
    }
}

__global__ void scatter_qkv(const uint16_t* __restrict__ qkv,
                            uint16_t* __restrict__ qb, uint16_t* __restrict__ kb,
                            uint16_t* __restrict__ vb)
{
    long long idx = (long long)blockIdx.x * 256 + threadIdx.x;
    if (idx >= (long long)ROWS * 3456) return;
    int m = (int)(idx / 3456), c = (int)(idx % 3456);
    int s = m >> 8, t = m & 255;
    int i = c / D_;  int rem = c - i * D_;
    int h = rem / HD_, d = rem - h * HD_;
    uint16_t val = qkv[idx];
    long long bidx = (long long)(s * HH + h);
    if (i == 0) {
        float f = bf2f(val) * 0.11785113f;    // 1/sqrt(72)
        qb[(bidx * T_ + t) * HDP + d] = f2bf(f);
    } else if (i == 1) {
        kb[(bidx * T_ + t) * HDP + d] = val;
    } else {
        vb[(bidx * HDP + d) * T_ + t] = val;  // V stored transposed [d][t]
    }
}

__global__ __launch_bounds__(128)
void softmax_rows(uint16_t* __restrict__ ps)   // 512*256 rows of length 256
{
    int row  = blockIdx.x * 4 + (threadIdx.x >> 5);
    int lane = threadIdx.x & 31;
    uint16_t* p = ps + (long long)row * 256;
    float v[8];
    float mx = -1e30f;
#pragma unroll
    for (int j = 0; j < 8; j++) { v[j] = bf2f(p[lane + j * 32]); mx = fmaxf(mx, v[j]); }
    mx = wave_max(mx);
    float sum = 0.f;
#pragma unroll
    for (int j = 0; j < 8; j++) { v[j] = __expf(v[j] - mx); sum += v[j]; }
    sum = wave_sum(sum);
    float inv = 1.f / sum;
#pragma unroll
    for (int j = 0; j < 8; j++) p[lane + j * 32] = f2bf(v[j] * inv);
}

__global__ void gather_o(const uint16_t* __restrict__ opad, uint16_t* __restrict__ attn_o)
{
    long long idx = (long long)blockIdx.x * 256 + threadIdx.x;
    if (idx >= (long long)ROWS * D_) return;
    int m = (int)(idx / D_), col = (int)(idx % D_);
    int s = m >> 8, t = m & 255;
    int h = col / HD_, d = col - h * HD_;
    attn_o[idx] = opad[((long long)(s * HH + h) * T_ + t) * HDP + d];
}

__global__ void residual_gate(float* __restrict__ xf, const float* __restrict__ c,
                              const float* __restrict__ mod, int gOff)
{
    long long idx = (long long)blockIdx.x * 256 + threadIdx.x;
    if (idx >= (long long)ROWS * D_) return;
    int m = (int)(idx / D_), d = (int)(idx % D_);
    int s = m >> 8;
    xf[idx] += mod[(long long)s * 6912 + gOff + d] * c[idx];
}

__global__ void gelu_bf(const float* __restrict__ in, uint16_t* __restrict__ out, long long n)
{
    long long i = (long long)blockIdx.x * 256 + threadIdx.x;
    if (i >= n) return;
    float x = in[i];
    float t = tanhf(0.7978845608028654f * (x + 0.044715f * x * x * x));
    out[i] = f2bf(0.5f * x * (1.f + t));
}

__global__ void pool_mod(const float* __restrict__ lnx, const float* __restrict__ mod_f,
                         uint16_t* __restrict__ pooled)
{
    int idx = blockIdx.x * 256 + threadIdx.x;
    if (idx >= NB * D_) return;
    int n = idx / D_, d = idx - n * D_;
    float s = 0.f;
    for (int t = 0; t < T_; t++) s += lnx[(long long)(n * T_ + t) * D_ + d];
    s *= (1.f / T_);
    float sh = mod_f[n * 2304 + d];
    float sc = mod_f[n * 2304 + 1152 + d];
    pooled[idx] = f2bf(s * (1.f + sc) + sh);
}

__global__ void silu_inplace(float* __restrict__ p, long long n)
{
    long long i = (long long)blockIdx.x * 256 + threadIdx.x;
    if (i < n) { float v = p[i]; p[i] = v / (1.f + __expf(-v)); }
}

__global__ __launch_bounds__(256)
void energy_kernel(const float* __restrict__ hid, const float* __restrict__ We2,
                   float* __restrict__ out)
{
    int n = blockIdx.x;
    float s = 0.f;
    for (int i = threadIdx.x; i < 2304; i += 256) s += hid[n * 2304 + i] * We2[i];
    s = wave_sum(s);
    __shared__ float sb[8];
    if ((threadIdx.x & 31) == 0) sb[threadIdx.x >> 5] = s;
    __syncthreads();
    if (threadIdx.x == 0) {
        float S = 0.f;
#pragma unroll
        for (int i = 0; i < 8; i++) S += sb[i];
        out[n] = S;
    }
}

// ---------------------------------------------------------------------------
extern "C" void kernel_launch(void* const* d_in, const int* in_sizes, int n_in,
                              void* d_out, int out_size, void* d_ws, size_t ws_size,
                              hipStream_t stream)
{
    (void)in_sizes; (void)n_in; (void)out_size; (void)ws_size;
    const float* x      = (const float*)d_in[0];
    const float* y      = (const float*)d_in[1];
    const float* pos    = (const float*)d_in[2];
    const float* Wqkv   = (const float*)d_in[3];
    const float* bqkv   = (const float*)d_in[4];
    const float* Wproj  = (const float*)d_in[5];
    const float* bproj  = (const float*)d_in[6];
    const float* W1     = (const float*)d_in[7];
    const float* b1     = (const float*)d_in[8];
    const float* W2     = (const float*)d_in[9];
    const float* b2     = (const float*)d_in[10];
    const float* Wada   = (const float*)d_in[11];
    const float* bada   = (const float*)d_in[12];
    const float* Wada_f = (const float*)d_in[13];
    const float* bada_f = (const float*)d_in[14];
    const float* We1    = (const float*)d_in[15];
    const float* be1    = (const float*)d_in[16];
    const float* We2    = (const float*)d_in[17];
    float* out = (float*)d_out;

    char* ws = (char*)d_ws;
    size_t off = 0;
    auto alloc = [&](size_t bytes) -> char* {
        char* p = ws + off;
        off = (off + bytes + 255) & ~(size_t)255;
        return p;
    };
    float*    xf     = (float*)   alloc((size_t)ROWS * D_ * 4);
    uint16_t* sc_bf  = (uint16_t*)alloc((size_t)NB * D_ * 2);
    float*    modb   = (float*)   alloc((size_t)NB * 6912 * 4);
    uint16_t* h_bf   = (uint16_t*)alloc((size_t)ROWS * D_ * 2);
    uint16_t* wstage = (uint16_t*)alloc((size_t)6912 * D_ * 2);
    uint16_t* qkv_c  = (uint16_t*)alloc((size_t)ROWS * 3456 * 2);
    uint16_t* qb     = (uint16_t*)alloc((size_t)512 * T_ * HDP * 2);
    uint16_t* kb     = (uint16_t*)alloc((size_t)512 * T_ * HDP * 2);
    uint16_t* vb     = (uint16_t*)alloc((size_t)512 * T_ * HDP * 2);
    uint16_t* ps     = (uint16_t*)alloc((size_t)512 * 256 * 256 * 2);
    uint16_t* attn_o = (uint16_t*)alloc((size_t)ROWS * D_ * 2);
    float*    cf32   = (float*)   alloc((size_t)ROWS * DF_ * 4);
    uint16_t* h1_bf  = (uint16_t*)alloc((size_t)ROWS * DF_ * 2);
    float*    mod_f  = (float*)   alloc((size_t)NB * 2304 * 4);
    uint16_t* pooled = (uint16_t*)alloc((size_t)NB * D_ * 2);
    float*    hid    = (float*)   alloc((size_t)NB * 2304 * 4);

    auto elw = [](long long n) { return dim3((unsigned)((n + 255) / 256)); };
    auto gemmGrid = [](int M, int N, int Z) {
        return dim3((unsigned)((N + 63) / 64), (unsigned)((M + 127) / 128), (unsigned)Z);
    };
    const long long nXD = (long long)ROWS * D_;

    add_pos_kernel<<<elw(nXD), 256, 0, stream>>>(x, pos, xf, nXD);
    silu_bf_kernel<<<elw(NB * D_), 256, 0, stream>>>(y, sc_bf, NB * D_);
    fill_zero_bf<<<elw(512LL * T_ * HDP), 256, 0, stream>>>(qb, 512LL * T_ * HDP);
    fill_zero_bf<<<elw(512LL * T_ * HDP), 256, 0, stream>>>(kb, 512LL * T_ * HDP);
    fill_zero_bf<<<elw(512LL * T_ * HDP), 256, 0, stream>>>(vb, 512LL * T_ * HDP);

    for (int l = 0; l < LL; l++) {
        // adaLN modulation: mod = silu(c) @ Wada^T + bada  -> [32, 6912]
        cvt_kernel<<<elw(6912LL * D_), 256, 0, stream>>>(Wada + (long long)l * 6912 * D_, wstage, 6912LL * D_);
        wmma_gemm<float><<<gemmGrid(NB, 6912, 1), 128, 0, stream>>>(
            sc_bf, D_, 0, wstage, D_, 0, modb, 6912, 0, bada + (long long)l * 6912, NB, 6912, D_);

        // attention branch
        ln_modulate<uint16_t><<<ROWS, 256, 0, stream>>>(xf, modb, 6912, 0 * D_, 1 * D_, h_bf);
        cvt_kernel<<<elw(3456LL * D_), 256, 0, stream>>>(Wqkv + (long long)l * 3456 * D_, wstage, 3456LL * D_);
        wmma_gemm<uint16_t><<<gemmGrid(ROWS, 3456, 1), 128, 0, stream>>>(
            h_bf, D_, 0, wstage, D_, 0, qkv_c, 3456, 0, bqkv + (long long)l * 3456, ROWS, 3456, D_);
        scatter_qkv<<<elw((long long)ROWS * 3456), 256, 0, stream>>>(qkv_c, qb, kb, vb);

        // S = q @ k^T  (batched over 512 (n,h))
        wmma_gemm<uint16_t><<<gemmGrid(T_, T_, 512), 128, 0, stream>>>(
            qb, HDP, (long long)T_ * HDP, kb, HDP, (long long)T_ * HDP,
            ps, 256, 65536LL, nullptr, 256, 256, HDP);
        softmax_rows<<<(512 * 256) / 4, 128, 0, stream>>>(ps);
        // O_pad = P @ V (V pre-transposed); write into qb (reused as o_pad)
        wmma_gemm<uint16_t><<<gemmGrid(T_, HDP, 512), 128, 0, stream>>>(
            ps, 256, 65536LL, vb, T_, (long long)HDP * T_,
            qb, HDP, (long long)T_ * HDP, nullptr, 256, HDP, 256);
        gather_o<<<elw(nXD), 256, 0, stream>>>(qb, attn_o);

        // proj + gated residual
        cvt_kernel<<<elw((long long)D_ * D_), 256, 0, stream>>>(Wproj + (long long)l * D_ * D_, wstage, (long long)D_ * D_);
        wmma_gemm<float><<<gemmGrid(ROWS, D_, 1), 128, 0, stream>>>(
            attn_o, D_, 0, wstage, D_, 0, cf32, D_, 0, bproj + (long long)l * D_, ROWS, D_, D_);
        residual_gate<<<elw(nXD), 256, 0, stream>>>(xf, cf32, modb, 2 * D_);

        // MLP branch
        ln_modulate<uint16_t><<<ROWS, 256, 0, stream>>>(xf, modb, 6912, 3 * D_, 4 * D_, h_bf);
        cvt_kernel<<<elw((long long)DF_ * D_), 256, 0, stream>>>(W1 + (long long)l * DF_ * D_, wstage, (long long)DF_ * D_);
        wmma_gemm<float><<<gemmGrid(ROWS, DF_, 1), 128, 0, stream>>>(
            h_bf, D_, 0, wstage, D_, 0, cf32, DF_, 0, b1 + (long long)l * DF_, ROWS, DF_, D_);
        gelu_bf<<<elw((long long)ROWS * DF_), 256, 0, stream>>>(cf32, h1_bf, (long long)ROWS * DF_);
        cvt_kernel<<<elw((long long)D_ * DF_), 256, 0, stream>>>(W2 + (long long)l * D_ * DF_, wstage, (long long)D_ * DF_);
        wmma_gemm<float><<<gemmGrid(ROWS, D_, 1), 128, 0, stream>>>(
            h1_bf, DF_, 0, wstage, DF_, 0, cf32, D_, 0, b2 + (long long)l * D_, ROWS, D_, DF_);
        residual_gate<<<elw(nXD), 256, 0, stream>>>(xf, cf32, modb, 5 * D_);
    }

    // final layer
    cvt_kernel<<<elw(2304LL * D_), 256, 0, stream>>>(Wada_f, wstage, 2304LL * D_);
    wmma_gemm<float><<<gemmGrid(NB, 2304, 1), 128, 0, stream>>>(
        sc_bf, D_, 0, wstage, D_, 0, mod_f, 2304, 0, bada_f, NB, 2304, D_);
    ln_modulate<float><<<ROWS, 256, 0, stream>>>(xf, (const float*)nullptr, 0, 0, 0, cf32);
    pool_mod<<<elw(NB * D_), 256, 0, stream>>>(cf32, mod_f, pooled);
    cvt_kernel<<<elw(2304LL * D_), 256, 0, stream>>>(We1, wstage, 2304LL * D_);
    wmma_gemm<float><<<gemmGrid(NB, 2304, 1), 128, 0, stream>>>(
        pooled, D_, 0, wstage, D_, 0, hid, 2304, 0, be1, NB, 2304, D_);
    silu_inplace<<<elw(NB * 2304), 256, 0, stream>>>(hid, NB * 2304);
    energy_kernel<<<NB, 256, 0, stream>>>(hid, We2, out);
}
